// Method_GCN_10393820856553
// MI455X (gfx1250) — matrix operand order repaired and checked by reading, not compiled, not run on hardware
//
#include <hip/hip_runtime.h>
#include <hip/hip_bf16.h>

// ---------------------------------------------------------------------------
// GCN forward for MI455X (gfx1250, wave32).
//   h1 = relu(Ahat @ (x @ W1) + b1)
//   out = log_softmax(Ahat @ (h1 @ W2) + b2)
// Ahat = D^-1/2 (A + I) D^-1/2, applied as per-edge scatter-add with weights
// dinv[row]*dinv[col]; self-loop handled analytically in finalize kernels.
// Dense GEMMs use v_wmma_f32_16x16x32_bf16 (16x16x32), scatter uses
// global_atomic_add_f32 (memory/atomic bound part of the workload).
// ---------------------------------------------------------------------------

typedef __attribute__((ext_vector_type(16))) __bf16 v16bf;
typedef __attribute__((ext_vector_type(8)))  __bf16 v8bf;
typedef __attribute__((ext_vector_type(8)))  float  v8f;

#define F_IN  128
#define HID   35
#define NCLS  10
#define HIDP  64   // padded K for layer-2 GEMM (2 x 32)
#define NCLSP 16   // padded N for layer-2 GEMM

// ---------------- weight pre-swizzle into WMMA B layout ----------------
// B layout (16-bit, 32x16 KxN): lane L in [0,15]  -> N = L,    K = 0..7, 16..23
//                               lane L in [16,31] -> N = L-16, K = 8..15, 24..31
// v16bf element j of lane L holds B[(j>>3)*16 + (L>>4)*8 + (j&7)][L&15].
__global__ void prep_weights(const float* __restrict__ W1,
                             const float* __restrict__ W2,
                             __bf16* __restrict__ W1s,
                             __bf16* __restrict__ W2s) {
    int t = threadIdx.x;
    // W1s: [3 ntiles][4 ksteps][32 lanes][16 elems]
    for (int idx = t; idx < 3 * 4 * 32 * 16; idx += 256) {
        int j  = idx & 15;
        int L  = (idx >> 4) & 31;
        int ks = (idx >> 9) & 3;
        int nt = idx >> 11;
        int k  = ks * 32 + (j >> 3) * 16 + (L >> 4) * 8 + (j & 7);   // 0..127, always valid
        int n  = nt * 16 + (L & 15);            // 0..47
        float v = (n < HID) ? W1[k * HID + n] : 0.0f;
        W1s[idx] = (__bf16)v;
    }
    // W2s: [2 ksteps][32 lanes][16 elems], K padded 35->64, N padded 10->16
    for (int idx = t; idx < 2 * 32 * 16; idx += 256) {
        int j  = idx & 15;
        int L  = (idx >> 4) & 31;
        int ks = idx >> 9;
        int k  = ks * 32 + (j >> 3) * 16 + (L >> 4) * 8 + (j & 7);   // 0..63
        int n  = L & 15;                        // 0..15
        float v = (k < HID && n < NCLS) ? W2[k * NCLS + n] : 0.0f;
        W2s[idx] = (__bf16)v;
    }
}

// ---------------- degree + normalization ----------------
__global__ void deg_kernel(const int* __restrict__ ei, float* __restrict__ deg,
                           int n_edges) {
    int e = blockIdx.x * blockDim.x + threadIdx.x;
    if (e < n_edges) {
        int col = ei[n_edges + e];
        atomicAdd(&deg[col], 1.0f);
    }
}

__global__ void dinv_kernel(float* __restrict__ deg, int n_nodes) {
    int i = blockIdx.x * blockDim.x + threadIdx.x;
    if (i < n_nodes) {
        float d = deg[i] + 1.0f;   // +1 self loop; always > 0
        deg[i] = rsqrtf(d);        // in place: deg -> dinv
    }
}

// ---------------- GEMM1: hlin1[N,35] = x[N,128] @ W1[128,35] ----------------
// One wave per 16-row tile; A loaded once (4 x v16bf), 3 N-tiles x 4 WMMAs.
__global__ void gemm1(const float* __restrict__ x,
                      const __bf16* __restrict__ W1s,
                      float* __restrict__ hlin1,
                      int n_rowblocks) {
    int wid  = threadIdx.x >> 5;
    int lane = threadIdx.x & 31;
    int rb   = blockIdx.x * 8 + wid;
    if (rb >= n_rowblocks) return;          // wave-uniform exit

    int khalf = lane >> 4;
    int r     = rb * 16 + (lane & 15);
    const float* xr = x + (size_t)r * F_IN;

    // A layout (16-bit, 16x32 MxK): lane holds K = khalf*8+0..7 and 16+khalf*8+0..7
    v16bf A[4];
#pragma unroll
    for (int ks = 0; ks < 4; ++ks) {
        int kb = ks * 32;
        v8f c1 = *(const v8f*)(xr + kb + khalf * 8);
        v8f c2 = *(const v8f*)(xr + kb + 16 + khalf * 8);
#pragma unroll
        for (int j = 0; j < 8; ++j) {
            A[ks][j]     = (__bf16)c1[j];
            A[ks][j + 8] = (__bf16)c2[j];
        }
    }

#pragma unroll
    for (int nt = 0; nt < 3; ++nt) {
        v8f acc = {};
#pragma unroll
        for (int ks = 0; ks < 4; ++ks) {
            v16bf B = *(const v16bf*)(W1s + (((size_t)(nt * 4 + ks) * 32 + lane) * 16));
            acc = __builtin_amdgcn_wmma_f32_16x16x32_bf16(
                false, A[ks], false, B, (short)0, acc, false, false);
        }
        int col = nt * 16 + (lane & 15);
        if (col < HID) {
            int rbase = rb * 16 + 8 * khalf;   // C/D: vgpr j -> row j (+8 hi half)
#pragma unroll
            for (int j = 0; j < 8; ++j)
                hlin1[(size_t)(rbase + j) * HID + col] = acc[j];
        }
    }
}

// ---------------- edge scatter, layer 1 (35 features) ----------------
__global__ void scatter1(const int* __restrict__ ei,
                         const float* __restrict__ dinv,
                         const float* __restrict__ hlin1,
                         float* __restrict__ agg1,
                         int n_edges) {
    int e = blockIdx.x * 8 + threadIdx.y;
    if (e >= n_edges) return;
    int row = ei[e];
    int col = ei[n_edges + e];
    float w = dinv[row] * dinv[col];
    const float* hsrc = hlin1 + (size_t)row * HID;
    float*       adst = agg1  + (size_t)col * HID;
    int f = threadIdx.x;                       // 0..31
    atomicAdd(&adst[f], hsrc[f] * w);
    if (f < HID - 32)
        atomicAdd(&adst[32 + f], hsrc[32 + f] * w);
}

// ---------------- finalize layer 1: +self loop, +b1, relu, -> bf16 padded ----
__global__ void finalize1(const float* __restrict__ agg1,
                          const float* __restrict__ hlin1,
                          const float* __restrict__ dinv,
                          const float* __restrict__ b1,
                          __bf16* __restrict__ h1bf,
                          int n_nodes) {
    int i = blockIdx.x * 4 + threadIdx.y;
    if (i >= n_nodes) return;
    int f = threadIdx.x;                       // 0..63
    float out = 0.0f;
    if (f < HID) {
        float di = dinv[i];
        float v = agg1[(size_t)i * HID + f]
                + di * di * hlin1[(size_t)i * HID + f]
                + b1[f];
        out = fmaxf(v, 0.0f);
    }
    h1bf[(size_t)i * HIDP + f] = (__bf16)out;  // cols 35..63 zero padded
}

// ---------------- GEMM2: hlin2[N,16] = h1bf[N,64] @ W2p[64,16] ----------------
__global__ void gemm2(const __bf16* __restrict__ h1bf,
                      const __bf16* __restrict__ W2s,
                      float* __restrict__ hlin2,
                      int n_rowblocks) {
    int wid  = threadIdx.x >> 5;
    int lane = threadIdx.x & 31;
    int rb   = blockIdx.x * 8 + wid;
    if (rb >= n_rowblocks) return;

    int khalf = lane >> 4;
    int r     = rb * 16 + (lane & 15);
    const __bf16* hr = h1bf + (size_t)r * HIDP;

    v16bf A[2];
#pragma unroll
    for (int ks = 0; ks < 2; ++ks) {
        v8bf c1 = *(const v8bf*)(hr + ks * 32 + khalf * 8);
        v8bf c2 = *(const v8bf*)(hr + ks * 32 + 16 + khalf * 8);
#pragma unroll
        for (int j = 0; j < 8; ++j) {
            A[ks][j]     = c1[j];
            A[ks][j + 8] = c2[j];
        }
    }

    v8f acc = {};
#pragma unroll
    for (int ks = 0; ks < 2; ++ks) {
        v16bf B = *(const v16bf*)(W2s + (((size_t)ks * 32 + lane) * 16));
        acc = __builtin_amdgcn_wmma_f32_16x16x32_bf16(
            false, A[ks], false, B, (short)0, acc, false, false);
    }

    int col   = lane & 15;
    int rbase = rb * 16 + 8 * khalf;
#pragma unroll
    for (int j = 0; j < 8; ++j)
        hlin2[(size_t)(rbase + j) * NCLSP + col] = acc[j];   // pad cols are 0
}

// ---------------- edge scatter, layer 2 (10 features) ----------------
__global__ void scatter2(const int* __restrict__ ei,
                         const float* __restrict__ dinv,
                         const float* __restrict__ hlin2,
                         float* __restrict__ agg2,
                         int n_edges) {
    int e = blockIdx.x * 16 + threadIdx.y;
    if (e >= n_edges) return;
    int f = threadIdx.x;                       // 0..15
    if (f >= NCLS) return;
    int row = ei[e];
    int col = ei[n_edges + e];
    float w = dinv[row] * dinv[col];
    atomicAdd(&agg2[(size_t)col * NCLSP + f],
              hlin2[(size_t)row * NCLSP + f] * w);
}

// ---------------- finalize layer 2: +self loop, +b2, log_softmax -------------
__global__ void finalize2(const float* __restrict__ agg2,
                          const float* __restrict__ hlin2,
                          const float* __restrict__ dinv,
                          const float* __restrict__ b2,
                          float* __restrict__ out,
                          int n_nodes) {
    int i = blockIdx.x * blockDim.x + threadIdx.x;
    if (i >= n_nodes) return;
    float di = dinv[i];
    float d2 = di * di;
    float v[NCLS];
    float m = -3.0e38f;
#pragma unroll
    for (int f = 0; f < NCLS; ++f) {
        float t = agg2[(size_t)i * NCLSP + f]
                + d2 * hlin2[(size_t)i * NCLSP + f]
                + b2[f];
        v[f] = t;
        m = fmaxf(m, t);
    }
    float s = 0.0f;
#pragma unroll
    for (int f = 0; f < NCLS; ++f) s += __expf(v[f] - m);
    float l = __logf(s);
#pragma unroll
    for (int f = 0; f < NCLS; ++f)
        out[(size_t)i * NCLS + f] = v[f] - m - l;
}

// ---------------------------------------------------------------------------
extern "C" void kernel_launch(void* const* d_in, const int* in_sizes, int n_in,
                              void* d_out, int out_size, void* d_ws, size_t ws_size,
                              hipStream_t stream) {
    const float* x  = (const float*)d_in[0];
    const int*   ei = (const int*)d_in[1];     // [2, E] flat
    const float* W1 = (const float*)d_in[2];
    const float* b1 = (const float*)d_in[3];
    const float* W2 = (const float*)d_in[4];
    const float* b2 = (const float*)d_in[5];
    float* out = (float*)d_out;

    const int N = in_sizes[0] / F_IN;          // 200000
    const int E = in_sizes[1] / 2;             // 3200000

    // workspace layout (byte offsets, 256B aligned for N=200000)
    char* ws = (char*)d_ws;
    const size_t OFF_DEG   = 0;                               // N f32
    const size_t OFF_HLIN1 = OFF_DEG   + (size_t)N * 4;       // N*35 f32
    const size_t OFF_AGG1  = OFF_HLIN1 + (size_t)N * HID * 4; // N*35 f32
    const size_t OFF_H1BF  = OFF_AGG1  + (size_t)N * HID * 4; // N*64 bf16
    const size_t OFF_HLIN2 = OFF_H1BF  + (size_t)N * HIDP * 2;// N*16 f32
    const size_t OFF_AGG2  = OFF_HLIN2 + (size_t)N * NCLSP * 4;// N*16 f32
    const size_t OFF_W1S   = OFF_AGG2  + (size_t)N * NCLSP * 4;// 6144 bf16
    const size_t OFF_W2S   = OFF_W1S   + 3 * 4 * 32 * 16 * 2;  // 1024 bf16

    float*  deg   = (float*)(ws + OFF_DEG);    // becomes dinv in place
    float*  hlin1 = (float*)(ws + OFF_HLIN1);
    float*  agg1  = (float*)(ws + OFF_AGG1);
    __bf16* h1bf  = (__bf16*)(ws + OFF_H1BF);
    float*  hlin2 = (float*)(ws + OFF_HLIN2);
    float*  agg2  = (float*)(ws + OFF_AGG2);
    __bf16* W1s   = (__bf16*)(ws + OFF_W1S);
    __bf16* W2s   = (__bf16*)(ws + OFF_W2S);

    // deterministic per call: clear accumulators (stream-ordered, capturable)
    hipMemsetAsync(deg,  0, (size_t)N * 4,           stream);
    hipMemsetAsync(agg1, 0, (size_t)N * HID * 4,     stream);
    hipMemsetAsync(agg2, 0, (size_t)N * NCLSP * 4,   stream);

    prep_weights<<<1, 256, 0, stream>>>(W1, W2, W1s, W2s);

    deg_kernel<<<(E + 255) / 256, 256, 0, stream>>>(ei, deg, E);
    dinv_kernel<<<(N + 255) / 256, 256, 0, stream>>>(deg, N);

    const int n_rowblocks = (N + 15) / 16;     // 12500
    const int gemm_grid   = (n_rowblocks + 7) / 8;
    gemm1<<<gemm_grid, 256, 0, stream>>>(x, W1s, hlin1, n_rowblocks);

    scatter1<<<(E + 7) / 8, dim3(32, 8), 0, stream>>>(ei, deg, hlin1, agg1, E);

    finalize1<<<(N + 3) / 4, dim3(64, 4), 0, stream>>>(agg1, hlin1, deg, b1, h1bf, N);

    gemm2<<<gemm_grid, 256, 0, stream>>>(h1bf, W2s, hlin2, n_rowblocks);

    scatter2<<<(E + 15) / 16, dim3(16, 16), 0, stream>>>(ei, deg, hlin2, agg2, E);

    finalize2<<<(N + 255) / 256, 256, 0, stream>>>(agg2, hlin2, deg, b2, out, N);
}
